// Block_24790551232950
// MI455X (gfx1250) — matrix operand order repaired and checked by reading, not compiled, not run on hardware
//
#include <hip/hip_runtime.h>
#include <hip/hip_bf16.h>
#include <cstdint>

typedef unsigned short u16;
typedef __attribute__((ext_vector_type(16))) __bf16 v16bf;
typedef __attribute__((ext_vector_type(8)))  float  v8f;

union FragU { uint4 q[2]; v16bf v; };

// ---- CDNA5 async global->LDS path (guarded; falls back to sync staging) ----
#if defined(__has_builtin)
#if __has_builtin(__builtin_amdgcn_global_load_async_to_lds_b128) && \
    __has_builtin(__builtin_amdgcn_s_wait_asynccnt)
#define HAVE_ASYNC_LDS 1
#endif
#endif
#ifndef HAVE_ASYNC_LDS
#define HAVE_ASYNC_LDS 0
#endif

#if HAVE_ASYNC_LDS
// Exact pointee type from the toolchain's builtin signature:
//   __attribute__((vector_size(4*sizeof(int)))) int __device__ *
typedef __attribute__((__vector_size__(4 * sizeof(int)))) int async_i4;
#define TO_GLOBAL(p) \
  ((__attribute__((address_space(1))) async_i4*)(unsigned long long)(uintptr_t)(p))
#define TO_LDS(p) \
  ((__attribute__((address_space(3))) async_i4*)(unsigned int)(uintptr_t)(p))
#endif

__device__ __forceinline__ void cp_b128_to_lds(u16* lds_dst, const u16* gsrc) {
#if HAVE_ASYNC_LDS
  __builtin_amdgcn_global_load_async_to_lds_b128(TO_GLOBAL(gsrc), TO_LDS(lds_dst), 0, 0);
#else
  *(uint4*)lds_dst = *(const uint4*)gsrc;
#endif
}
__device__ __forceinline__ void async_wait_all() {
#if HAVE_ASYNC_LDS
  __builtin_amdgcn_s_wait_asynccnt(0);
#endif
}
__device__ __forceinline__ void async_wait_4() {
#if HAVE_ASYNC_LDS
  __builtin_amdgcn_s_wait_asynccnt(4);
#endif
}

__device__ __forceinline__ u16 f32_to_bf16(float f) {
  unsigned int u = __float_as_uint(f);
  unsigned int r = u + 0x7FFFu + ((u >> 16) & 1u);
  return (u16)(r >> 16);
}

// A-matrix fragment (16x32 bf16). ISA layout: lane L(0-15): row M=L, K=[0..8)+[16..24);
// lane L+16: row M=L, K=[8..16)+[24..32). Two 16B chunks per lane.
__device__ __forceinline__ v16bf load_frag_a(const u16* base, int ld) {
  const int lane = threadIdx.x & 31;
  const u16* p = base + (size_t)(lane & 15) * (size_t)ld + ((lane >> 4) << 3);
  FragU u;
  u.q[0] = *(const uint4*)(p);
  u.q[1] = *(const uint4*)(p + 16);
  return u.v;
}

// B-matrix fragment (32x16 bf16), source stored column-major ([N,K] row-major).
// Lane L(0-15): col N=L, K=[0..8)+[16..24); lane L+16: col N=L, K=[8..16)+[24..32).
__device__ __forceinline__ v16bf load_frag_b(const u16* base, int ld) {
  const int lane = threadIdx.x & 31;
  const u16* p = base + (size_t)(lane & 15) * (size_t)ld + ((lane >> 4) << 3);
  FragU u;
  u.q[0] = *(const uint4*)(p);
  u.q[1] = *(const uint4*)(p + 16);
  return u.v;
}

__device__ __forceinline__ v8f wmma_bf16(v16bf a, v16bf b, v8f c) {
  return __builtin_amdgcn_wmma_f32_16x16x32_bf16(false, a, false, b, (short)0, c,
                                                 false, false);
}

__device__ __forceinline__ float hmax16(float v) {
  v = fmaxf(v, __shfl_xor(v, 1, 32));
  v = fmaxf(v, __shfl_xor(v, 2, 32));
  v = fmaxf(v, __shfl_xor(v, 4, 32));
  v = fmaxf(v, __shfl_xor(v, 8, 32));
  return v;
}
__device__ __forceinline__ float hsum16(float v) {
  v += __shfl_xor(v, 1, 32);
  v += __shfl_xor(v, 2, 32);
  v += __shfl_xor(v, 4, 32);
  v += __shfl_xor(v, 8, 32);
  return v;
}

// ---------------------------------------------------------------------------
// Weight transpose + f32->bf16: Wt[n*K + k] = bf16(W[k*N + n])
// ---------------------------------------------------------------------------
__global__ void wt_convert_kernel(const float* __restrict__ W, u16* __restrict__ Wt,
                                  int K, int N) {
  size_t idx = (size_t)blockIdx.x * 256 + threadIdx.x;
  if (idx >= (size_t)K * (size_t)N) return;
  size_t n = idx / (size_t)K;
  size_t k = idx % (size_t)K;
  Wt[idx] = f32_to_bf16(W[k * (size_t)N + n]);
}

// ---------------------------------------------------------------------------
// LayerNorm (one row per block) -> bf16
// ---------------------------------------------------------------------------
__global__ __launch_bounds__(256) void ln_bf16_kernel(
    const float* __restrict__ x, const float* __restrict__ g,
    const float* __restrict__ b, u16* __restrict__ out, int D) {
  const int row = blockIdx.x;
  const int tid = threadIdx.x;
  const float* xr = x + (size_t)row * D;
  float s = 0.f, ss = 0.f;
  for (int i = tid; i < D; i += 256) { float v = xr[i]; s += v; ss += v * v; }
  for (int m = 1; m < 32; m <<= 1) { s += __shfl_xor(s, m, 32); ss += __shfl_xor(ss, m, 32); }
  __shared__ float reds[8], redss[8];
  const int wave = tid >> 5, lane = tid & 31;
  if (lane == 0) { reds[wave] = s; redss[wave] = ss; }
  __syncthreads();
  float S = 0.f, SS = 0.f;
  for (int w = 0; w < 8; w++) { S += reds[w]; SS += redss[w]; }
  const float invD = 1.0f / (float)D;
  const float mu = S * invD;
  const float var = SS * invD - mu * mu;
  const float inv = rsqrtf(var + 1e-5f);
  for (int i = tid; i < D; i += 256)
    out[(size_t)row * D + i] = f32_to_bf16((xr[i] - mu) * inv * g[i] + b[i]);
}

// ---------------------------------------------------------------------------
// LDS-staged, double-buffered WMMA GEMM:
//   C[M,N] = A[M,K](bf16) * Wt[N,K](bf16)^T + bias
// Block tile 128x128, K-step 32. Per K-step the 256 threads DMA the 128x32
// A-tile and 128x32 B-tile into LDS (async global->LDS when available), all
// 8 waves then read fragments from LDS (padded stride 40 -> conflict-free
// ds_load_b128). Wave tile 32x64 = 2x4 WMMA accumulators.
// Epilogues:
//  EPI 0: f32 out + residual      EPI 1: exact GELU -> bf16 row-major
//  EPI 2: bf16 scatter [B,H,T,HD] EPI 3: bf16 scatter [B,H,HD,T] (V transpose)
// ---------------------------------------------------------------------------
#define LDS_STRIDE 40  // 32 data + 8 pad halfwords (80B rows: conflict-free b128)

template <int EPI>
__global__ __launch_bounds__(256) void gemm_bf16_kernel(
    const u16* __restrict__ A, const u16* __restrict__ Wt,
    const float* __restrict__ bias, const float* __restrict__ resid,
    float* __restrict__ outf, u16* __restrict__ outb, int M, int N, int K) {
  const int tid = threadIdx.x;
  const int lane = tid & 31;
  const int wave = tid >> 5;
  const int waveM = wave >> 1;         // 0..3
  const int waveN = wave & 1;          // 0..1
  const int mBase = blockIdx.y * 128;
  const int nBase = blockIdx.x * 128;
  const int rowBase = mBase + waveM * 32;
  const int colBase = nBase + waveN * 64;

  __shared__ __align__(16) u16 Abuf[2][128 * LDS_STRIDE];
  __shared__ __align__(16) u16 Bbuf[2][128 * LDS_STRIDE];

  // Per-thread staging slots: 2 x 16B chunks of A, 2 of B per K-step.
  const int cid0 = tid, cid1 = tid + 256;       // chunk ids, row = cid>>2, col = (cid&3)*8
  const int ar0 = cid0 >> 2, ac0 = (cid0 & 3) * 8;
  const int ar1 = cid1 >> 2, ac1 = (cid1 & 3) * 8;

  const u16* gA = A + (size_t)mBase * K;
  const u16* gB = Wt + (size_t)nBase * K;

  v8f acc[2][4];
  for (int i = 0; i < 2; i++)
    for (int j = 0; j < 4; j++)
      for (int e = 0; e < 8; e++) acc[i][j][e] = 0.f;

  const int KT = K >> 5;  // number of 32-wide K steps

  // Prologue: stage K-step 0 into buffer 0.
  cp_b128_to_lds(&Abuf[0][ar0 * LDS_STRIDE + ac0], gA + (size_t)ar0 * K + ac0);
  cp_b128_to_lds(&Abuf[0][ar1 * LDS_STRIDE + ac1], gA + (size_t)ar1 * K + ac1);
  cp_b128_to_lds(&Bbuf[0][ar0 * LDS_STRIDE + ac0], gB + (size_t)ar0 * K + ac0);
  cp_b128_to_lds(&Bbuf[0][ar1 * LDS_STRIDE + ac1], gB + (size_t)ar1 * K + ac1);

  for (int kt = 0; kt < KT; kt++) {
    const int cur = kt & 1;
    const int k_next = (kt + 1) << 5;
    if (kt + 1 < KT) {
      const int nb = cur ^ 1;
      cp_b128_to_lds(&Abuf[nb][ar0 * LDS_STRIDE + ac0], gA + (size_t)ar0 * K + k_next + ac0);
      cp_b128_to_lds(&Abuf[nb][ar1 * LDS_STRIDE + ac1], gA + (size_t)ar1 * K + k_next + ac1);
      cp_b128_to_lds(&Bbuf[nb][ar0 * LDS_STRIDE + ac0], gB + (size_t)ar0 * K + k_next + ac0);
      cp_b128_to_lds(&Bbuf[nb][ar1 * LDS_STRIDE + ac1], gB + (size_t)ar1 * K + k_next + ac1);
      async_wait_4();   // current buffer's 4 DMAs complete; next 4 in flight
    } else {
      async_wait_all();
    }
    __syncthreads();    // all threads' staging of `cur` visible

    v16bf fa[2], fb[4];
#pragma unroll
    for (int i = 0; i < 2; i++)
      fa[i] = load_frag_a(&Abuf[cur][(waveM * 32 + 16 * i) * LDS_STRIDE], LDS_STRIDE);
#pragma unroll
    for (int j = 0; j < 4; j++)
      fb[j] = load_frag_b(&Bbuf[cur][(waveN * 64 + 16 * j) * LDS_STRIDE], LDS_STRIDE);
#pragma unroll
    for (int i = 0; i < 2; i++)
#pragma unroll
      for (int j = 0; j < 4; j++)
        acc[i][j] = wmma_bf16(fa[i], fb[j], acc[i][j]);

    __syncthreads();    // all waves done reading `cur` before it is re-staged
  }

  const int hsel = lane >> 4;          // C layout: M = r + hsel*8
  const int nlane = lane & 15;         // C layout: N = nlane
#pragma unroll
  for (int i = 0; i < 2; i++)
#pragma unroll
    for (int j = 0; j < 4; j++)
#pragma unroll
      for (int r = 0; r < 8; r++) {
        const int m = rowBase + 16 * i + hsel * 8 + r;
        const int n = colBase + 16 * j + nlane;
        float v = acc[i][j][r] + bias[n];
        if (EPI == 0) {
          outf[(size_t)m * N + n] = v + resid[(size_t)m * N + n];
        } else if (EPI == 1) {
          float ge = 0.5f * v * (1.0f + erff(v * 0.70710678118654752f));
          outb[(size_t)m * N + n] = f32_to_bf16(ge);
        } else if (EPI == 2) {  // q/k: [B,H,T,HD], T=2048, H=16, HD=64
          const int bb = m >> 11, t = m & 2047;
          const int h = n >> 6, hd = n & 63;
          outb[(((size_t)(bb * 16 + h)) * 2048 + t) * 64 + hd] = f32_to_bf16(v);
        } else {               // v: [B,H,HD,T]
          const int bb = m >> 11, t = m & 2047;
          const int h = n >> 6, hd = n & 63;
          outb[(((size_t)(bb * 16 + h)) * 64 + hd) * 2048 + t] = f32_to_bf16(v);
        }
      }
}

// ---------------------------------------------------------------------------
// Flash attention. Grid (T/64, B*H), 128 threads (4 waves x 16 q-rows).
// S = Q K^T via 2 WMMAs/ntile; online softmax with half-wave reductions;
// P transposed via LDS into A-fragment layout; O += P V via WMMA.
// ---------------------------------------------------------------------------
__global__ __launch_bounds__(128) void attn_kernel(
    const u16* __restrict__ Qb, const u16* __restrict__ Kb,
    const u16* __restrict__ Vtb, u16* __restrict__ Yb) {
  constexpr int T = 2048, HD = 64, D = 1024;
  const int bh = blockIdx.y;
  const int qTile = blockIdx.x;
  const int wave = threadIdx.x >> 5;
  const int lane = threadIdx.x & 31;
  const int nlane = lane & 15, hsel = lane >> 4;
  const int qBase = qTile * 64 + wave * 16;

  const u16* Q = Qb + (size_t)bh * T * HD;
  const u16* Kp = Kb + (size_t)bh * T * HD;
  const u16* Vt = Vtb + (size_t)bh * HD * T;

  v16bf qf[2];
  qf[0] = load_frag_a(Q + (size_t)qBase * HD, HD);
  qf[1] = load_frag_a(Q + (size_t)qBase * HD + 32, HD);

  v8f o[4];
  for (int j = 0; j < 4; j++)
    for (int e = 0; e < 8; e++) o[j][e] = 0.f;
  float mrow[8], lrow[8];
  for (int r = 0; r < 8; r++) { mrow[r] = -INFINITY; lrow[r] = 0.f; }

  __shared__ __align__(16) u16 pT[4][16 * 32];
  const float scale = 0.125f;  // 1/sqrt(64)

  const int kEnd = qTile * 64 + 64;
  for (int j0 = 0; j0 < kEnd; j0 += 32) {
    // ---- S = Q K^T for 16 rows x 32 keys
    v8f s[2];
#pragma unroll
    for (int nt = 0; nt < 2; nt++) {
      for (int e = 0; e < 8; e++) s[nt][e] = 0.f;
      v16bf kf0 = load_frag_b(Kp + (size_t)(j0 + nt * 16) * HD, HD);
      v16bf kf1 = load_frag_b(Kp + (size_t)(j0 + nt * 16) * HD + 32, HD);
      s[nt] = wmma_bf16(qf[0], kf0, s[nt]);
      s[nt] = wmma_bf16(qf[1], kf1, s[nt]);
    }
    // ---- scale + causal mask
#pragma unroll
    for (int nt = 0; nt < 2; nt++) {
      const int jcol = j0 + nt * 16 + nlane;
#pragma unroll
      for (int r = 0; r < 8; r++) {
        const int qrow = qBase + hsel * 8 + r;
        const float v = s[nt][r] * scale;
        s[nt][r] = (jcol <= qrow) ? v : -INFINITY;
      }
    }
    // ---- online softmax
    float alpha[8];
#pragma unroll
    for (int r = 0; r < 8; r++) {
      const float rm = hmax16(fmaxf(s[0][r], s[1][r]));
      const float mn = fmaxf(mrow[r], rm);
      alpha[r] = __expf(mrow[r] - mn);
      mrow[r] = mn;
    }
#pragma unroll
    for (int r = 0; r < 8; r++) {
      const float p0 = __expf(s[0][r] - mrow[r]);
      const float p1 = __expf(s[1][r] - mrow[r]);
      const int prow = hsel * 8 + r;
      pT[wave][prow * 32 + nlane] = f32_to_bf16(p0);
      pT[wave][prow * 32 + 16 + nlane] = f32_to_bf16(p1);
      lrow[r] = lrow[r] * alpha[r] + hsum16(p0 + p1);
    }
#pragma unroll
    for (int j = 0; j < 4; j++)
#pragma unroll
      for (int r = 0; r < 8; r++) o[j][r] *= alpha[r];
    // ---- P fragment from LDS (A layout: row nlane, chunks hsel*8 and 16+hsel*8)
    v16bf pf;
    {
      const u16* p = &pT[wave][nlane * 32 + hsel * 8];
      FragU u;
      u.q[0] = *(const uint4*)(p);
      u.q[1] = *(const uint4*)(p + 16);
      pf = u.v;
    }
    // ---- O += P V  (V pre-transposed: [HD, T])
#pragma unroll
    for (int j = 0; j < 4; j++) {
      v16bf vf = load_frag_b(Vt + (size_t)(j * 16) * T + j0, T);
      o[j] = wmma_bf16(pf, vf, o[j]);
    }
  }

  const int bb = bh >> 4, h = bh & 15;
#pragma unroll
  for (int j = 0; j < 4; j++)
#pragma unroll
    for (int r = 0; r < 8; r++) {
      const int q = qBase + hsel * 8 + r;
      const int hd = j * 16 + nlane;
      const float val = o[j][r] / lrow[r];
      Yb[((size_t)(bb * T + q)) * D + h * 64 + hd] = f32_to_bf16(val);
    }
}

// ---------------------------------------------------------------------------
extern "C" void kernel_launch(void* const* d_in, const int* in_sizes, int n_in,
                              void* d_out, int out_size, void* d_ws, size_t ws_size,
                              hipStream_t stream) {
  (void)in_sizes; (void)n_in; (void)out_size; (void)ws_size;
  const float* x    = (const float*)d_in[0];
  const float* ln1g = (const float*)d_in[1];
  const float* ln1b = (const float*)d_in[2];
  const float* ln2g = (const float*)d_in[3];
  const float* ln2b = (const float*)d_in[4];
  const float* Wq = (const float*)d_in[5];  const float* bq = (const float*)d_in[6];
  const float* Wk = (const float*)d_in[7];  const float* bk = (const float*)d_in[8];
  const float* Wv = (const float*)d_in[9];  const float* bv = (const float*)d_in[10];
  const float* Wo = (const float*)d_in[11]; const float* bo = (const float*)d_in[12];
  const float* W1 = (const float*)d_in[13]; const float* b1 = (const float*)d_in[14];
  const float* W2 = (const float*)d_in[15]; const float* b2 = (const float*)d_in[16];

  constexpr int Bn = 4, T = 2048, D = 1024, H = 16, DFF = 4096;
  constexpr int M = Bn * T;  // 8192

  char* ws = (char*)d_ws;
  size_t off = 0;
  auto alloc = [&](size_t bytes) {
    char* p = ws + off;
    off += (bytes + 255) & ~(size_t)255;
    return p;
  };
  u16* hbf  = (u16*)alloc((size_t)M * D * 2);      // LN1 out (bf16)
  u16* WqT  = (u16*)alloc((size_t)D * D * 2);
  u16* WkT  = (u16*)alloc((size_t)D * D * 2);
  u16* WvT  = (u16*)alloc((size_t)D * D * 2);
  u16* WoT  = (u16*)alloc((size_t)D * D * 2);
  u16* W1T  = (u16*)alloc((size_t)D * DFF * 2);
  u16* W2T  = (u16*)alloc((size_t)DFF * D * 2);
  u16* Qbf  = (u16*)alloc((size_t)M * D * 2);      // [B,H,T,HD]
  u16* Kbf  = (u16*)alloc((size_t)M * D * 2);      // [B,H,T,HD]
  u16* Vtbf = (u16*)alloc((size_t)M * D * 2);      // [B,H,HD,T]
  u16* ybf  = (u16*)alloc((size_t)M * D * 2);      // attention out (bf16)
  float* x1 = (float*)alloc((size_t)M * D * 4);    // x + attn (f32)
  u16* h2bf = (u16*)alloc((size_t)M * D * 2);      // LN2 out
  u16* mbf  = (u16*)alloc((size_t)M * DFF * 2);    // gelu(mlp1) (bf16)

  const dim3 blk(256);
  // Weight casts/transposes
  wt_convert_kernel<<<(D * D + 255) / 256, blk, 0, stream>>>(Wq, WqT, D, D);
  wt_convert_kernel<<<(D * D + 255) / 256, blk, 0, stream>>>(Wk, WkT, D, D);
  wt_convert_kernel<<<(D * D + 255) / 256, blk, 0, stream>>>(Wv, WvT, D, D);
  wt_convert_kernel<<<(D * D + 255) / 256, blk, 0, stream>>>(Wo, WoT, D, D);
  wt_convert_kernel<<<(D * DFF + 255) / 256, blk, 0, stream>>>(W1, W1T, D, DFF);
  wt_convert_kernel<<<(DFF * D + 255) / 256, blk, 0, stream>>>(W2, W2T, DFF, D);

  // LN1
  ln_bf16_kernel<<<M, blk, 0, stream>>>(x, ln1g, ln1b, hbf, D);

  // QKV projections
  const dim3 gDD(D / 128, M / 128);
  gemm_bf16_kernel<2><<<gDD, blk, 0, stream>>>(hbf, WqT, bq, nullptr, nullptr, Qbf, M, D, D);
  gemm_bf16_kernel<2><<<gDD, blk, 0, stream>>>(hbf, WkT, bk, nullptr, nullptr, Kbf, M, D, D);
  gemm_bf16_kernel<3><<<gDD, blk, 0, stream>>>(hbf, WvT, bv, nullptr, nullptr, Vtbf, M, D, D);

  // Attention
  attn_kernel<<<dim3(T / 64, Bn * H), dim3(128), 0, stream>>>(Qbf, Kbf, Vtbf, ybf);

  // Output projection + residual
  gemm_bf16_kernel<0><<<gDD, blk, 0, stream>>>(ybf, WoT, bo, x, x1, nullptr, M, D, D);

  // LN2 + MLP
  ln_bf16_kernel<<<M, blk, 0, stream>>>(x1, ln2g, ln2b, h2bf, D);
  gemm_bf16_kernel<1><<<dim3(DFF / 128, M / 128), blk, 0, stream>>>(
      h2bf, W1T, b1, nullptr, nullptr, mbf, M, DFF, D);
  gemm_bf16_kernel<0><<<gDD, blk, 0, stream>>>(
      mbf, W2T, b2, x1, (float*)d_out, nullptr, M, D, DFF);
}
